// CustomTransformer_80968723464437
// MI455X (gfx1250) — compile-verified, matrix operand order checked by reading, not run on hardware
//
#include <hip/hip_runtime.h>
#include <hip/hip_bf16.h>
#include <math.h>

// ---------------------------------------------------------------------------
// Problem constants: B=2, S=1024, D=1024, H=16 (DH=64), FF=4096, V=32000, L=6
// ---------------------------------------------------------------------------
#define BB   2
#define SS   1024
#define DDim 1024
#define HH   16
#define DH   64
#define FFd  4096
#define VV   32000
#define LL   6

typedef __attribute__((ext_vector_type(16))) _Float16 v16h;
typedef __attribute__((ext_vector_type(8)))  _Float16 v8h;
typedef __attribute__((ext_vector_type(8)))  float    v8f;
typedef __attribute__((ext_vector_type(4)))  int      v4i;

// ---------------------------------------------------------------------------
// CDNA5 async global->LDS copy (ASYNCcnt path) with portable fallback.
// Builtin signature (probed via diagnostic): (v4i AS1*, v4i AS3*, imm, imm)
// ---------------------------------------------------------------------------
#if defined(__has_builtin)
#if __has_builtin(__builtin_amdgcn_global_load_async_to_lds_b128)
#define HAS_ASYNC_LDS 1
#endif
#endif
#ifndef HAS_ASYNC_LDS
#define HAS_ASYNC_LDS 0
#endif

__device__ __forceinline__ void cp16_g2l(const _Float16* g, _Float16* l) {
#if HAS_ASYNC_LDS
    __builtin_amdgcn_global_load_async_to_lds_b128(
        (__attribute__((address_space(1))) v4i*)(g),
        (__attribute__((address_space(3))) v4i*)(l), 0, 0);
#else
    *(v8h*)l = *(const v8h*)g;
#endif
}

__device__ __forceinline__ void wait_async_lds() {
#if HAS_ASYNC_LDS
#if __has_builtin(__builtin_amdgcn_s_wait_asynccnt)
    __builtin_amdgcn_s_wait_asynccnt(0);
#else
    asm volatile("s_wait_asynccnt 0" ::: "memory");
#endif
#endif
}

// D = A(16x32 f16) * B(32x16 f16) + C(16x16 f32)
__device__ __forceinline__ v8f wmma16(v16h a, v16h b, v8f c) {
    return __builtin_amdgcn_wmma_f32_16x16x32_f16(
        false, a, false, b, (short)0, c, false, false);
}

// A-fragment (16x32, MxK) per CDNA5 ISA layout:
//   lanes 0-15 : row M=lane,    elems 0..7 = K0..7,  8..15 = K16..23
//   lanes 16-31: row M=lane-16, elems 0..7 = K8..15, 8..15 = K24..31
// p points at (row = lane&15, K = (lane>>4)*8); K contiguous in memory.
__device__ __forceinline__ v16h make_frag(const _Float16* p) {
    v8h lo = *(const v8h*)(p);
    v8h hi = *(const v8h*)(p + 16);
    return __builtin_shufflevector(lo, hi, 0,1,2,3,4,5,6,7,8,9,10,11,12,13,14,15);
}

__device__ __forceinline__ v16h frag_a_lds(const _Float16* tile, int ld) {
    const int lane = threadIdx.x & 31;
    return make_frag(tile + (lane & 15) * ld + ((lane >> 4) << 3));
}

// B-fragment from row-major [K][N] storage (scalar path; used for V in attn).
__device__ __forceinline__ v16h frag_b_rowmajor(const _Float16* base, int ld) {
    const int lane = threadIdx.x & 31;
    const _Float16* col = base + (lane & 15);
    const int k0 = (lane >> 4) << 3;
    v16h r;
#pragma unroll
    for (int j = 0; j < 8; ++j) r[j]     = col[(k0 + j) * ld];
#pragma unroll
    for (int j = 0; j < 8; ++j) r[8 + j] = col[(16 + k0 + j) * ld];
    return r;
}

// ---------------------------------------------------------------------------
// Embedding + sinusoidal PE: h = emb[x]*sqrt(D) + PE  (fp32 + f16 copies)
// ---------------------------------------------------------------------------
__global__ __launch_bounds__(256) void embed_kernel(
    const long long* __restrict__ x, const float* __restrict__ emb,
    float* __restrict__ h, _Float16* __restrict__ h16) {
    const int token = blockIdx.x;
    const int pos   = token % SS;
    const long long id = x[token];
    const float* erow = emb + (size_t)id * DDim;
    const size_t base = (size_t)token * DDim;
#pragma unroll
    for (int i = 0; i < 4; ++i) {
        const int d = threadIdx.x + 256 * i;
        const int even = d & ~1;
        const float div = __expf((float)even * (-9.210340372f / (float)DDim));
        const float ang = (float)pos * div;
        const float pe  = (d & 1) ? __cosf(ang) : __sinf(ang);
        const float v   = erow[d] * 32.0f + pe;   // sqrt(1024)=32
        h[base + d]   = v;
        h16[base + d] = (_Float16)v;
    }
}

// ---------------------------------------------------------------------------
// C = act(A16[M,K] @ W[K,N] + bias).  A in f16 (async-copied to LDS),
// W fp32 (converted & transposed into LDS).  Optional fp32 and/or f16 output.
// Block 128x128, K-step 32, 8 waves (2x4), wave tile 64x32 (4x2 WMMA).
// ---------------------------------------------------------------------------
#define BM 128
#define BN 128
#define BK 32

__global__ __launch_bounds__(256) void gemm_f16_wmma(
    const _Float16* __restrict__ A, const float* __restrict__ W,
    const float* __restrict__ bias, float* __restrict__ Cf,
    _Float16* __restrict__ Ch, int M, int N, int K, int act) {
    __shared__ _Float16 As[BM * BK];    // row-major [m][k]
    __shared__ _Float16 BsT[BN * BK];   // column-major [n][k]

    const int tid   = threadIdx.x;
    const int lane  = tid & 31;
    const int wave  = tid >> 5;
    const int waveM = wave >> 2;   // 0..1
    const int waveN = wave & 3;    // 0..3
    const int m0    = blockIdx.y * BM;
    const int n0    = blockIdx.x * BN;
    const int k0l   = (lane >> 4) << 3;

    const v8f zero8 = {0.f, 0.f, 0.f, 0.f, 0.f, 0.f, 0.f, 0.f};
    v8f acc[4][2];
#pragma unroll
    for (int i = 0; i < 4; ++i)
#pragma unroll
        for (int j = 0; j < 2; ++j) acc[i][j] = zero8;

    // Staging assignments (fixed per thread)
    const int arow = tid >> 1;            // A: 0..127
    const int acol = (tid & 1) * 16;      // A: halves 0 or 16
    const int bn   = tid & 127;           // B: column within tile
    const int bk   = (tid >> 7) * 16;     // B: k-range 0..15 or 16..31

    for (int k0 = 0; k0 < K; k0 += BK) {
        __syncthreads();
        // --- A tile: f16 global -> LDS (async path on CDNA5) ---
        {
            const _Float16* g = A + (size_t)(m0 + arow) * K + k0 + acol;
            _Float16* l = As + arow * BK + acol;
            cp16_g2l(g, l);
            cp16_g2l(g + 8, l + 8);
        }
        // --- B tile: fp32 global -> f16 LDS, transposed to column-major ---
        {
            const float* g = W + (size_t)(k0 + bk) * N + n0 + bn;
            float t[16];
#pragma unroll
            for (int i = 0; i < 16; ++i) t[i] = g[(size_t)i * N];
            v8h p0, p1;
#pragma unroll
            for (int i = 0; i < 8; ++i) { p0[i] = (_Float16)t[i]; p1[i] = (_Float16)t[8 + i]; }
            _Float16* l = BsT + bn * BK + bk;
            *(v8h*)l = p0;
            *(v8h*)(l + 8) = p1;
        }
        if (k0 + BK < K) {   // global_prefetch_b8 for next K tile
            __builtin_prefetch(A + (size_t)(m0 + arow) * K + k0 + BK, 0, 1);
            __builtin_prefetch(W + (size_t)(k0 + BK + bk) * N + n0 + bn, 0, 1);
        }
        wait_async_lds();
        __syncthreads();

        v16h bf[2];
#pragma unroll
        for (int j = 0; j < 2; ++j)
            bf[j] = make_frag(BsT + (waveN * 32 + j * 16 + (lane & 15)) * BK + k0l);
#pragma unroll
        for (int i = 0; i < 4; ++i) {
            const v16h af = frag_a_lds(As + (waveM * 64 + i * 16) * BK, BK);
#pragma unroll
            for (int j = 0; j < 2; ++j)
                acc[i][j] = wmma16(af, bf[j], acc[i][j]);
        }
    }

    // Epilogue: element r of lane l is (M = r + 8*(lane>>4), N = lane&15)
    const int rowOff = 8 * (lane >> 4);
    const int nloc   = lane & 15;
#pragma unroll
    for (int i = 0; i < 4; ++i) {
#pragma unroll
        for (int j = 0; j < 2; ++j) {
            const int col = n0 + waveN * 32 + j * 16 + nloc;
            const float bv = bias[col];
#pragma unroll
            for (int r = 0; r < 8; ++r) {
                const int row = m0 + waveM * 64 + i * 16 + r + rowOff;
                float v = acc[i][j][r] + bv;
                if (act == 1) v = 0.5f * v * (1.0f + erff(v * 0.70710678f));
                if (Cf) Cf[(size_t)row * N + col] = v;
                if (Ch) Ch[(size_t)row * N + col] = (_Float16)v;
            }
        }
    }
}

// ---------------------------------------------------------------------------
// Flash-style causal attention, all operands f16 in [B,S,D] head-interleaved
// layout.  grid = (S/64, B*H), 128 threads = 4 waves, wave owns 16 q rows.
// Key tiles of 32; K/V async-staged to LDS; scores scaled by 1/sqrt(DH).
// ---------------------------------------------------------------------------
__global__ __launch_bounds__(128) void attn_kernel(
    const _Float16* __restrict__ Qg, const _Float16* __restrict__ Kg,
    const _Float16* __restrict__ Vg, _Float16* __restrict__ Og) {
    const int bh     = blockIdx.y;
    const int b      = bh / HH;
    const int head   = bh % HH;
    const int qblock = blockIdx.x * 64;
    const int wave   = threadIdx.x >> 5;
    const int lane   = threadIdx.x & 31;
    const int qrow0  = qblock + wave * 16;

    __shared__ _Float16 Ks[32 * DH];
    __shared__ _Float16 Vs[32 * DH];
    __shared__ _Float16 Ps[4][16 * 32];

    const size_t baseBH = (size_t)b * SS * DDim + (size_t)head * DH;

    // Q fragments: direct 16B global loads (layout already fragment-friendly)
    v16h qf[2];
    {
        const _Float16* qrow =
            Qg + baseBH + (size_t)(qrow0 + (lane & 15)) * DDim + ((lane >> 4) << 3);
        qf[0] = make_frag(qrow);
        qf[1] = make_frag(qrow + 32);
    }

    const v8f zero8 = {0.f, 0.f, 0.f, 0.f, 0.f, 0.f, 0.f, 0.f};
    v8f out[4];
#pragma unroll
    for (int c = 0; c < 4; ++c) out[c] = zero8;
    float mrow[8], lrow[8];
#pragma unroll
    for (int r = 0; r < 8; ++r) { mrow[r] = -1e30f; lrow[r] = 0.f; }

    const int ntiles = (qblock + 64) >> 5;
    for (int t = 0; t < ntiles; ++t) {
        const int kbase = t * 32;
        __syncthreads();
        // Async stage of K,V tiles (32 x 64 halves = 4KB each)
#pragma unroll
        for (int i = 0; i < 2; ++i) {
            const int c   = threadIdx.x * 2 + i;   // 0..255 chunks of 16B
            const int key = c >> 3;
            const int cc  = (c & 7) * 8;
            const size_t g = baseBH + (size_t)(kbase + key) * DDim + cc;
            cp16_g2l(Kg + g, Ks + key * DH + cc);
            cp16_g2l(Vg + g, Vs + key * DH + cc);
        }
        wait_async_lds();
        __syncthreads();

        if (kbase <= qrow0 + 15) {
            // scores 16x32 as two 16x16 tiles, accumulated over DH halves
            v8f s0 = zero8, s1 = zero8;
#pragma unroll
            for (int hd = 0; hd < 2; ++hd) {
                const int k0 = (lane >> 4) << 3;
                const v16h kb0 = make_frag(Ks + ((lane & 15)) * DH + hd * 32 + k0);
                const v16h kb1 = make_frag(Ks + ((lane & 15) + 16) * DH + hd * 32 + k0);
                s0 = wmma16(qf[hd], kb0, s0);
                s1 = wmma16(qf[hd], kb1, s1);
            }
            s0 = s0 * 0.125f;   // 1/sqrt(64)
            s1 = s1 * 0.125f;
            // causal mask
            const int qOff = qrow0 + 8 * (lane >> 4);
            const int key0 = kbase + (lane & 15);
#pragma unroll
            for (int r = 0; r < 8; ++r) {
                const int qi = qOff + r;
                if (key0 > qi)      s0[r] = -1e30f;
                if (key0 + 16 > qi) s1[r] = -1e30f;
            }
            // online softmax (row lives in one lane-half; reduce over 16 lanes)
            float alph[8];
#pragma unroll
            for (int r = 0; r < 8; ++r) {
                float rm = fmaxf(s0[r], s1[r]);
#pragma unroll
                for (int off = 8; off >= 1; off >>= 1) rm = fmaxf(rm, __shfl_xor(rm, off));
                const float nm = fmaxf(mrow[r], rm);
                alph[r] = __expf(mrow[r] - nm);
                mrow[r] = nm;
                const float p0 = __expf(s0[r] - nm);
                const float p1 = __expf(s1[r] - nm);
                float rs = p0 + p1;
#pragma unroll
                for (int off = 8; off >= 1; off >>= 1) rs += __shfl_xor(rs, off);
                lrow[r] = lrow[r] * alph[r] + rs;
                s0[r] = p0; s1[r] = p1;
            }
#pragma unroll
            for (int c = 0; c < 4; ++c)
#pragma unroll
                for (int r = 0; r < 8; ++r) out[c][r] *= alph[r];

            // P: C-layout -> wave-private LDS -> A-layout fragment
            _Float16* Pw = &Ps[wave][0];
#pragma unroll
            for (int r = 0; r < 8; ++r) {
                const int row = r + 8 * (lane >> 4);
                Pw[row * 32 + (lane & 15)]      = (_Float16)s0[r];
                Pw[row * 32 + 16 + (lane & 15)] = (_Float16)s1[r];
            }
            asm volatile("s_wait_dscnt 0" ::: "memory");
            const v16h pf = frag_a_lds(Pw, 32);
#pragma unroll
            for (int c = 0; c < 4; ++c) {
                const v16h vf = frag_b_rowmajor(Vs + c * 16, DH);
                out[c] = wmma16(pf, vf, out[c]);
            }
        }
    }

    float invl[8];
#pragma unroll
    for (int r = 0; r < 8; ++r) invl[r] = 1.0f / lrow[r];
#pragma unroll
    for (int c = 0; c < 4; ++c)
#pragma unroll
        for (int r = 0; r < 8; ++r) {
            const int row = qrow0 + r + 8 * (lane >> 4);
            const int col = c * 16 + (lane & 15);
            Og[baseBH + (size_t)row * DDim + col] = (_Float16)(out[c][r] * invl[r]);
        }
}

// ---------------------------------------------------------------------------
// out = LayerNorm(X + Y)*g + be  (fp32), also emits f16 copy for matmuls.
// ---------------------------------------------------------------------------
__global__ __launch_bounds__(256) void add_ln_kernel(
    const float* __restrict__ X, const float* __restrict__ Y,
    const float* __restrict__ g, const float* __restrict__ be,
    float* __restrict__ O, _Float16* __restrict__ O16) {
    __shared__ float red[8];
    const int tid = threadIdx.x;
    const size_t base = (size_t)blockIdx.x * DDim;

    float x[4]; float s = 0.f;
#pragma unroll
    for (int i = 0; i < 4; ++i) {
        const int d = tid + 256 * i;
        x[i] = X[base + d] + Y[base + d];
        s += x[i];
    }
#pragma unroll
    for (int off = 16; off >= 1; off >>= 1) s += __shfl_xor(s, off);
    if ((tid & 31) == 0) red[tid >> 5] = s;
    __syncthreads();
    float tot = 0.f;
#pragma unroll
    for (int w = 0; w < 8; ++w) tot += red[w];
    const float mean = tot * (1.0f / DDim);
    __syncthreads();

    float s2 = 0.f;
#pragma unroll
    for (int i = 0; i < 4; ++i) { const float dv = x[i] - mean; s2 += dv * dv; }
#pragma unroll
    for (int off = 16; off >= 1; off >>= 1) s2 += __shfl_xor(s2, off);
    if ((tid & 31) == 0) red[tid >> 5] = s2;
    __syncthreads();
    float var = 0.f;
#pragma unroll
    for (int w = 0; w < 8; ++w) var += red[w];
    const float rstd = rsqrtf(var * (1.0f / DDim) + 1e-5f);
#pragma unroll
    for (int i = 0; i < 4; ++i) {
        const int d = tid + 256 * i;
        const float v = (x[i] - mean) * rstd * g[d] + be[d];
        O[base + d]   = v;
        O16[base + d] = (_Float16)v;
    }
}

// ---------------------------------------------------------------------------
static inline void launch_gemm(const _Float16* A, const float* W, const float* bias,
                               float* Cf, _Float16* Ch, int M, int N, int K,
                               int act, hipStream_t s) {
    dim3 grid(N / BN, M / BM);
    gemm_f16_wmma<<<grid, 256, 0, s>>>(A, W, bias, Cf, Ch, M, N, K, act);
}

extern "C" void kernel_launch(void* const* d_in, const int* in_sizes, int n_in,
                              void* d_out, int out_size, void* d_ws, size_t ws_size,
                              hipStream_t stream) {
    const long long* x = (const long long*)d_in[0];
    // d_in[1] = causal mask (computed analytically)
    const float* emb  = (const float*)d_in[2];
    const float* Wq   = (const float*)d_in[3];
    const float* bq   = (const float*)d_in[4];
    const float* Wk   = (const float*)d_in[5];
    const float* bk   = (const float*)d_in[6];
    const float* Wv   = (const float*)d_in[7];
    const float* bv   = (const float*)d_in[8];
    const float* Wo   = (const float*)d_in[9];
    const float* bo   = (const float*)d_in[10];
    const float* g1   = (const float*)d_in[11];
    const float* be1  = (const float*)d_in[12];
    const float* g2   = (const float*)d_in[13];
    const float* be2  = (const float*)d_in[14];
    const float* W1   = (const float*)d_in[15];
    const float* b1   = (const float*)d_in[16];
    const float* W2   = (const float*)d_in[17];
    const float* b2   = (const float*)d_in[18];
    const float* Wout = (const float*)d_in[19];
    const float* bout = (const float*)d_in[20];
    float* logits = (float*)d_out;

    const int M = BB * SS;                 // 2048 tokens
    const size_t MD = (size_t)M * DDim;

    float*    h     = (float*)d_ws;        // fp32 residual stream
    float*    tmp   = h + MD;              // fp32 projection output
    _Float16* h16   = (_Float16*)(tmp + MD);
    _Float16* q16   = h16 + MD;
    _Float16* k16   = q16 + MD;
    _Float16* v16   = k16 + MD;
    _Float16* ctx16 = v16 + MD;
    _Float16* ffb16 = ctx16 + MD;          // 2048*4096 f16

    embed_kernel<<<M, 256, 0, stream>>>(x, emb, h, h16);

    for (int l = 0; l < LL; ++l) {
        const size_t dd  = (size_t)l * DDim * DDim;
        const size_t dv  = (size_t)l * DDim;
        const size_t df  = (size_t)l * DDim * FFd;
        const size_t dfv = (size_t)l * FFd;

        launch_gemm(h16, Wq + dd, bq + dv, nullptr, q16, M, DDim, DDim, 0, stream);
        launch_gemm(h16, Wk + dd, bk + dv, nullptr, k16, M, DDim, DDim, 0, stream);
        launch_gemm(h16, Wv + dd, bv + dv, nullptr, v16, M, DDim, DDim, 0, stream);

        attn_kernel<<<dim3(SS / 64, BB * HH), 128, 0, stream>>>(q16, k16, v16, ctx16);

        launch_gemm(ctx16, Wo + dd, bo + dv, tmp, nullptr, M, DDim, DDim, 0, stream);
        add_ln_kernel<<<M, 256, 0, stream>>>(h, tmp, g1 + dv, be1 + dv, h, h16);

        launch_gemm(h16, W1 + df, b1 + dfv, nullptr, ffb16, M, FFd, DDim, 1, stream);
        launch_gemm(ffb16, W2 + df, b2 + dv, tmp, nullptr, M, DDim, FFd, 0, stream);
        add_ln_kernel<<<M, 256, 0, stream>>>(h, tmp, g2 + dv, be2 + dv, h, h16);
    }

    launch_gemm(h16, Wout, bout, logits, nullptr, M, VV, DDim, 0, stream);
}
